// Encoder1_53266184405489
// MI455X (gfx1250) — matrix operand order repaired and checked by reading, not compiled, not run on hardware
//
#include <hip/hip_runtime.h>
#include <hip/hip_bf16.h>

// ---------------------------------------------------------------------------
// Bidirectional 2-layer LSTM, S=1024, D=H=1024.
// Phase A (parallel): Gx[t] = Wih @ x_t + bih + bhh  as bf16 WMMA GEMM.
// Phase B (sequential): g = Gx[t] + Whh @ h_{t-1}; gates; c,h update.
//   64 single-wave WGs per direction, 16 h-rows each, c held in registers,
//   per-timestep global arrival counters for the recurrence barrier.
// ---------------------------------------------------------------------------

typedef __attribute__((ext_vector_type(16))) __bf16 bf16x16;
typedef __attribute__((ext_vector_type(8)))  __bf16 bf16x8;
typedef __attribute__((ext_vector_type(8)))  float  f32x8;

#define SEQ   1024
#define HDIM  1024
#define G4    4096   // 4 * HDIM
#define NWG   64     // workgroups per direction in recurrence (64 * 16 rows = 1024)

__device__ __forceinline__ float fsigmoid(float x) {
    return 1.0f / (1.0f + __expf(-x));
}
__device__ __forceinline__ float ftanhf(float x) {
    float e = __expf(-2.0f * x);
    return (1.0f - e) / (1.0f + e);
}
__device__ __forceinline__ bf16x16 joinA(bf16x8 lo, bf16x8 hi) {
    return __builtin_shufflevector(lo, hi, 0,1,2,3,4,5,6,7,8,9,10,11,12,13,14,15);
}

// ---------------------------------------------------------------------------
__global__ void k_zero_i32(int* p, int n) {
    int i = blockIdx.x * blockDim.x + threadIdx.x;
    if (i < n) p[i] = 0;
}

// embedding gather + f32->bf16
__global__ void k_gather_bf16(const int* __restrict__ tok,
                              const float* __restrict__ emb,
                              __bf16* __restrict__ xs) {
    int t = blockIdx.x;
    const float* src = emb + (size_t)tok[t] * HDIM;
    __bf16* dst = xs + (size_t)t * HDIM;
    for (int d = threadIdx.x; d < HDIM; d += blockDim.x)
        dst[d] = (__bf16)src[d];
}

// f32 -> bf16 weight convert (grid-stride)
__global__ void k_cvt_bf16(const float* __restrict__ src,
                           __bf16* __restrict__ dst, int n) {
    int i = blockIdx.x * blockDim.x + threadIdx.x;
    int stride = gridDim.x * blockDim.x;
    for (; i < n; i += stride) dst[i] = (__bf16)src[i];
}

// ---------------------------------------------------------------------------
// Gx[dir][s][r] = sum_k W[r][k] * X[map(s)][k] + bih[r] + bhh[r]
// W: [4096][1024] bf16 row-major. X: [1024][1024] bf16.
// blockDim=128 (4 waves, 4 adjacent 16-row tiles), grid (64, 64, 2).
// WMMA 16x16x32 bf16: A = weight tile, B columns = 16 consecutive timesteps.
__global__ __launch_bounds__(128)
void k_gemm_gx(const __bf16* __restrict__ WA, const __bf16* __restrict__ WB,
               const __bf16* __restrict__ XA, const __bf16* __restrict__ XB,
               const float* __restrict__ bihA, const float* __restrict__ bhhA,
               const float* __restrict__ bihB, const float* __restrict__ bhhB,
               float* __restrict__ Gx, int revB)
{
    const int dir = blockIdx.z;
    const __bf16* W   = dir ? WB   : WA;
    const __bf16* X   = dir ? XB   : XA;
    const float*  bih = dir ? bihB : bihA;
    const float*  bhh = dir ? bhhB : bhhA;
    const int     rev = dir ? revB : 0;
    float* G = Gx + (size_t)dir * SEQ * G4;

    const int lane = threadIdx.x & 31;
    const int wv   = threadIdx.x >> 5;
    const int half = lane >> 4;
    const int l16  = lane & 15;

    const int r0 = (blockIdx.x * 4 + wv) * 16;   // output-row tile
    const int s  = blockIdx.y * 16 + l16;        // processing-order step (B column N=l16)
    const int trow = rev ? (SEQ - 1 - s) : s;    // fold time reversal into addressing

    // A 16-bit 16x32 layout: lane l holds row r0+l16,
    // K = half*8 + {0..7} and half*8 + 16 + {0..7}
    const __bf16* wrow = W + (size_t)(r0 + l16) * HDIM;
    // B 16-bit 32x16 layout: lane l holds N=l16, K = half*8 + {0..7} and half*8 + 16 + {0..7}
    const __bf16* xrow = X + (size_t)trow * HDIM;

    f32x8 acc = {};
    for (int kk = 0; kk < HDIM; kk += 32) {
        bf16x8 alo = *(const bf16x8*)(wrow + kk + half * 8);
        bf16x8 ahi = *(const bf16x8*)(wrow + kk + 16 + half * 8);
        bf16x16 a  = joinA(alo, ahi);
        bf16x16 b  = joinA(*(const bf16x8*)(xrow + kk + half * 8), *(const bf16x8*)(xrow + kk + 16 + half * 8));
        acc = __builtin_amdgcn_wmma_f32_16x16x32_bf16(
                  false, a, false, b, (short)0, acc, false, false);
    }

    // C/D layout: VGPR q = row (half*8 + q), column N = l16 -> step s
    const int rbase = r0 + half * 8;
    float* gout = G + (size_t)s * G4 + rbase;
    #pragma unroll
    for (int q = 0; q < 8; ++q)
        gout[q] = acc[q] + bih[rbase + q] + bhh[rbase + q];
}

// ---------------------------------------------------------------------------
// Sequential recurrence for one layer, both directions concurrently.
// grid = 128 blocks x 32 threads (1 wave). dir = blk>>6, wg = blk&63.
// Each WG owns h rows [wg*16, wg*16+16) and all 4 gate rows for them.
__global__ __launch_bounds__(32)
void k_recur(const __bf16* __restrict__ WhhA, const __bf16* __restrict__ WhhB,
             const float* __restrict__ Gx,     // [2][SEQ][4096]
             __bf16* __restrict__ h_all,       // [2][SEQ][1024] (bf16, feeds next step / next layer)
             float* __restrict__ out_final,    // [SEQ][2048] or nullptr
             float* __restrict__ out_hidden,   // [2048] row for this layer
             float* __restrict__ out_cell,     // [2048] row for this layer
             int* __restrict__ sync,           // [2][SEQ] arrival counters (pre-zeroed)
             int write_final)
{
    const int dir  = blockIdx.x >> 6;
    const int wg   = blockIdx.x & (NWG - 1);
    const int lane = threadIdx.x;
    const int half = lane >> 4;
    const int l16  = lane & 15;
    const int j0   = wg * 16;

    const __bf16* Whh = dir ? WhhB : WhhA;
    const float*  G   = Gx    + (size_t)dir * SEQ * G4;
    __bf16*       hA  = h_all + (size_t)dir * SEQ * HDIM;
    int*          sc  = sync  + dir * SEQ;

    __shared__ __bf16 hbuf[HDIM];

    // A-matrix weight rows for the 4 gates (row = gate*H + j0 + l16)
    const __bf16* w0 = Whh + (size_t)(0 * HDIM + j0 + l16) * HDIM;
    const __bf16* w1 = Whh + (size_t)(1 * HDIM + j0 + l16) * HDIM;
    const __bf16* w2 = Whh + (size_t)(2 * HDIM + j0 + l16) * HDIM;
    const __bf16* w3 = Whh + (size_t)(3 * HDIM + j0 + l16) * HDIM;

    float c[8];
    #pragma unroll
    for (int q = 0; q < 8; ++q) c[q] = 0.0f;   // c_{-1} = 0

    for (int t = 0; t < SEQ; ++t) {
        f32x8 a0 = {}, a1 = {}, a2 = {}, a3 = {};

        if (t > 0) {
            // wait for all 64 WGs of this direction to publish h_{t-1}
            while (__hip_atomic_load(&sc[t - 1], __ATOMIC_ACQUIRE,
                                     __HIP_MEMORY_SCOPE_AGENT) < NWG) {
                __builtin_amdgcn_s_sleep(1);
            }
            // stage h_{t-1} (2 KB bf16) into LDS
            const __bf16* hsrc = hA + (size_t)(t - 1) * HDIM;
            for (int i = lane; i < (HDIM * 2) / 16; i += 32)
                ((float4*)hbuf)[i] = ((const float4*)hsrc)[i];
#if __has_builtin(__builtin_amdgcn_s_wait_dscnt)
            __builtin_amdgcn_s_wait_dscnt(0);
#else
            __syncthreads();
#endif
            // Whh @ h_{t-1}: B operand broadcasts h across all 16 N columns,
            // so each lane holds the dot products of rows half*8 + {0..7}.
            for (int kk = 0; kk < HDIM; kk += 32) {
                bf16x16 b = joinA(*(const bf16x8*)(hbuf + kk + half * 8), *(const bf16x8*)(hbuf + kk + 16 + half * 8));
                if (kk + 32 < HDIM)
                    __builtin_prefetch(w0 + kk + 32, 0, 1);
                bf16x8 lo, hi;
                lo = *(const bf16x8*)(w0 + kk + half * 8);
                hi = *(const bf16x8*)(w0 + kk + 16 + half * 8);
                a0 = __builtin_amdgcn_wmma_f32_16x16x32_bf16(
                         false, joinA(lo, hi), false, b, (short)0, a0, false, false);
                lo = *(const bf16x8*)(w1 + kk + half * 8);
                hi = *(const bf16x8*)(w1 + kk + 16 + half * 8);
                a1 = __builtin_amdgcn_wmma_f32_16x16x32_bf16(
                         false, joinA(lo, hi), false, b, (short)0, a1, false, false);
                lo = *(const bf16x8*)(w2 + kk + half * 8);
                hi = *(const bf16x8*)(w2 + kk + 16 + half * 8);
                a2 = __builtin_amdgcn_wmma_f32_16x16x32_bf16(
                         false, joinA(lo, hi), false, b, (short)0, a2, false, false);
                lo = *(const bf16x8*)(w3 + kk + half * 8);
                hi = *(const bf16x8*)(w3 + kk + 16 + half * 8);
                a3 = __builtin_amdgcn_wmma_f32_16x16x32_bf16(
                         false, joinA(lo, hi), false, b, (short)0, a3, false, false);
            }
        }

        // gate nonlinearity + state update; lane holds rows j0+half*8+q
        const float* g = G + (size_t)t * G4;
        const int jb = j0 + half * 8;
        float hq[8];
        #pragma unroll
        for (int q = 0; q < 8; ++q) {
            int j = jb + q;
            float iv = fsigmoid(a0[q] + g[0 * HDIM + j]);
            float fv = fsigmoid(a1[q] + g[1 * HDIM + j]);
            float gv = ftanhf (a2[q] + g[2 * HDIM + j]);
            float ov = fsigmoid(a3[q] + g[3 * HDIM + j]);
            c[q]  = fv * c[q] + iv * gv;
            hq[q] = ov * ftanhf(c[q]);
        }

        // values are replicated across lanes within each half; lanes 0/16 publish
        if (l16 == 0) {
            __bf16* hd = hA + (size_t)t * HDIM + jb;
            #pragma unroll
            for (int q = 0; q < 8; ++q) hd[q] = (__bf16)hq[q];
            if (write_final) {
                float* of = dir ? (out_final + (size_t)(SEQ - 1 - t) * 2048 + 1024 + jb)
                                : (out_final + (size_t)t * 2048 + jb);
                #pragma unroll
                for (int q = 0; q < 8; ++q) of[q] = hq[q];
            }
            if (t == SEQ - 1) {
                #pragma unroll
                for (int q = 0; q < 8; ++q) {
                    out_hidden[dir * HDIM + jb + q] = hq[q];
                    out_cell  [dir * HDIM + jb + q] = c[q];
                }
            }
        }
        __threadfence();
        if (lane == 0)
            __hip_atomic_fetch_add(&sc[t], 1, __ATOMIC_RELEASE,
                                   __HIP_MEMORY_SCOPE_AGENT);
    }
}

// ---------------------------------------------------------------------------
extern "C" void kernel_launch(void* const* d_in, const int* in_sizes, int n_in,
                              void* d_out, int out_size, void* d_ws, size_t ws_size,
                              hipStream_t stream) {
    const int*   tok   = (const int*)  d_in[0];
    const float* emb   = (const float*)d_in[1];
    const float* Wih_f = (const float*)d_in[2];
    const float* Whh_f = (const float*)d_in[3];
    const float* bih_f = (const float*)d_in[4];
    const float* bhh_f = (const float*)d_in[5];
    const float* Wih_b = (const float*)d_in[6];
    const float* Whh_b = (const float*)d_in[7];
    const float* bih_b = (const float*)d_in[8];
    const float* bhh_b = (const float*)d_in[9];

    float* out        = (float*)d_out;
    float* out_cell   = out;          // [2][2048]
    float* out_hidden = out + 4096;   // [2][2048]
    float* out_final  = out + 8192;   // [SEQ][2048]

    char* ws = (char*)d_ws;
    size_t o = 0;
    __bf16* xs   = (__bf16*)(ws + o); o += (size_t)SEQ * HDIM * 2;          //  2 MiB
    __bf16* wihf = (__bf16*)(ws + o); o += (size_t)2 * G4 * HDIM * 2;       // 16 MiB
    __bf16* wihb = (__bf16*)(ws + o); o += (size_t)2 * G4 * HDIM * 2;       // 16 MiB
    __bf16* whhf = (__bf16*)(ws + o); o += (size_t)2 * G4 * HDIM * 2;       // 16 MiB
    __bf16* whhb = (__bf16*)(ws + o); o += (size_t)2 * G4 * HDIM * 2;       // 16 MiB
    float*  gx   = (float*) (ws + o); o += (size_t)2 * SEQ * G4 * 4;        // 32 MiB
    __bf16* h0   = (__bf16*)(ws + o); o += (size_t)2 * SEQ * HDIM * 2;      //  4 MiB
    __bf16* h1   = (__bf16*)(ws + o); o += (size_t)2 * SEQ * HDIM * 2;      //  4 MiB
    int*    sync = (int*)   (ws + o); o += (size_t)2 * 2 * SEQ * 4;         // 16 KiB

    const int nsync = 2 * 2 * SEQ;
    k_zero_i32<<<(nsync + 255) / 256, 256, 0, stream>>>(sync, nsync);
    k_gather_bf16<<<SEQ, 256, 0, stream>>>(tok, emb, xs);

    const int nW = 2 * G4 * HDIM;   // elements per (Wih or Whh) tensor
    k_cvt_bf16<<<4096, 256, 0, stream>>>(Wih_f, wihf, nW);
    k_cvt_bf16<<<4096, 256, 0, stream>>>(Wih_b, wihb, nW);
    k_cvt_bf16<<<4096, 256, 0, stream>>>(Whh_f, whhf, nW);
    k_cvt_bf16<<<4096, 256, 0, stream>>>(Whh_b, whhb, nW);

    dim3 gg(64, 64, 2);
    // ---- layer 0 ----
    k_gemm_gx<<<gg, 128, 0, stream>>>(wihf, wihb, xs, xs,
                                      bih_f, bhh_f, bih_b, bhh_b,
                                      gx, /*revB=*/1);
    k_recur<<<2 * NWG, 32, 0, stream>>>(whhf, whhb, gx, h0, nullptr,
                                        out_hidden + 0 * 2048, out_cell + 0 * 2048,
                                        sync, /*write_final=*/0);
    // ---- layer 1 (inputs are layer-0 h's, already in processing order) ----
    const size_t L1 = (size_t)G4 * HDIM;
    k_gemm_gx<<<gg, 128, 0, stream>>>(wihf + L1, wihb + L1,
                                      h0, h0 + (size_t)SEQ * HDIM,
                                      bih_f + G4, bhh_f + G4, bih_b + G4, bhh_b + G4,
                                      gx, /*revB=*/0);
    k_recur<<<2 * NWG, 32, 0, stream>>>(whhf + L1, whhb + L1, gx, h1, out_final,
                                        out_hidden + 1 * 2048, out_cell + 1 * 2048,
                                        sync + 2 * SEQ, /*write_final=*/1);
}
